// TensorNet_17626545783696
// MI455X (gfx1250) — compile-verified, matrix operand order checked by reading, not run on hardware
//
#include <hip/hip_runtime.h>
#include <hip/hip_bf16.h>

// ---------------------------------------------------------------------------
// TensorNet fused pipeline for MI455X (gfx1250, wave32, WMMA)
//   Dominant op: [128 x 4096] @ [4096 x 3*4096] GEMM streaming 201MB f32
//   weights once (23.3 TB/s -> ~8.6us floor). Split-K (8-way) x 3 slabs x
//   64 n-tiles = 1536 blocks / 12288 waves for occupancy; f32->f16
//   in-register conversion feeding v_wmma_f32_16x16x32_f16; double-buffered
//   LDS (1 barrier/chunk, 4 WMMA/wave/chunk); split-K partials combined with
//   global_atomic_add_f32; SELU epilogue in a separate pass.
// ---------------------------------------------------------------------------

typedef __attribute__((ext_vector_type(16))) _Float16 v16h;
typedef __attribute__((ext_vector_type(8)))  float    v8f;

static constexpr int ND     = 4096;
static constexpr int NNZ    = 110592;
static constexpr int BATCH  = 128;
static constexpr int NTILE  = 64;                 // columns per block
static constexpr int KSPLIT = 8;                  // K-splits per (ntile, slab)
static constexpr int KRANGE = ND / KSPLIT;        // 512
static constexpr int NCHUNK = KRANGE / 32;        // 16 K-chunks per block

// --------------------------- kernel 0: zero scratch ------------------------
__global__ void tn_zero(float4* __restrict__ p, int n4) {
    int i = blockIdx.x * blockDim.x + threadIdx.x;
    if (i < n4) p[i] = make_float4(0.f, 0.f, 0.f, 0.f);
}

// ------------------- kernel 1: transpose x -> xT[m][b] ---------------------
__global__ void tn_transpose(const float* __restrict__ x, float* __restrict__ xT) {
    int idx = blockIdx.x * blockDim.x + threadIdx.x;      // 524288 threads
    int b = idx >> 12;
    int m = idx & (ND - 1);
    xT[(size_t)m * BATCH + b] = x[idx];                   // coalesced reads
}

// ---------------- kernel 2: COO SpMV, one wave32 per nonzero ---------------
__global__ void tn_spmv(const int*   __restrict__ rows,
                        const int*   __restrict__ cols,
                        const float* __restrict__ vals,
                        const float* __restrict__ xT,
                        float*       __restrict__ yT) {
    int wid  = (blockIdx.x * blockDim.x + threadIdx.x) >> 5;
    int lane = threadIdx.x & 31;
    if (wid >= NNZ) return;
    int   r = rows[wid];
    int   c = cols[wid];
    float v = vals[wid];
    float4 xv = *(const float4*)(xT + (size_t)c * BATCH + lane * 4);
    float* yp = yT + (size_t)r * BATCH + lane * 4;
    atomicAdd(yp + 0, v * xv.x);
    atomicAdd(yp + 1, v * xv.y);
    atomicAdd(yp + 2, v * xv.z);
    atomicAdd(yp + 3, v * xv.w);
}

// -------- kernel 3: x1 = bc + yT*flag ; write as f16 A-matrix [b][m] -------
__global__ void tn_prep(const float* __restrict__ yT,
                        const float* __restrict__ bc,
                        const float* __restrict__ flag,
                        _Float16*    __restrict__ xh) {
    int idx = blockIdx.x * blockDim.x + threadIdx.x;      // m*128 + b
    int m = idx >> 7;
    int b = idx & (BATCH - 1);
    float v = bc[m] + yT[idx] * flag[m];
    xh[(size_t)b * ND + m] = (_Float16)v;
}

// --------------- kernel 4: split-K WMMA GEMM into tbuf (f32) ---------------
// Grid: 64 n-tiles x 3 slabs x 8 k-splits = 1536 blocks, 256 thr = 8 wave32.
// Each wave owns a 16-row batch tile; 4 WMMAs per wave per 32-K chunk.
__global__ __launch_bounds__(256) void tn_gemm_splitk(
        const _Float16* __restrict__ xh,    // [128][4096] f16 (A)
        const float*    __restrict__ w1,    // [3][4096][4096] f32
        float*          __restrict__ tbuf)  // [3][128][4096] f32 partials
{
    __shared__ alignas(16) char ldsB[2][4096];  // double-buffered 64x32 f16

    const int t    = threadIdx.x;
    const int lane = t & 31;
    const int wave = t >> 5;
    const int bid  = blockIdx.x;
    const int nt   = bid & 63;          // n-tile
    const int kk   = (bid >> 6) % 3;    // weight slab
    const int ks   = bid / 192;         // k-split
    const int n0    = nt * NTILE;
    const int kbase = ks * KRANGE;
    const int Mbase = wave * 16;

    // ---- producer: thread t loads 8 consecutive f32 weights for column pn,
    //      converts to f16, ds_store_b128 into exact B-fragment slot.
    const int pn    = t >> 2;                       // 0..63 column in tile
    const int g     = t & 3;                        // K group-of-8
    const int psub  = pn >> 4;                      // 16-wide N sub-tile
    const int plane = (pn & 15) + 16 * (g & 1);
    const int pofs  = psub * 1024 + plane * 32 + (g >> 1) * 16;
    const float* wrow = w1 + (size_t)kk * ND * ND + (size_t)(n0 + pn) * ND
                           + kbase + g * 8;

    // ---- consumer: A-fragment straight from global (row-major f16).
    const int mrow = Mbase + (lane & 15);
    const int kA0  = (lane < 16) ? 0 : 8;
    const _Float16* arow = xh + (size_t)mrow * ND + kbase;

    auto produce = [&](int buf, int c) {
        float4 a = *(const float4*)(wrow + c * 32);       // global_load_b128
        float4 b = *(const float4*)(wrow + c * 32 + 4);   // global_load_b128
        union { _Float16 h[8]; int4 u; } p;
        p.h[0] = (_Float16)a.x;  p.h[1] = (_Float16)a.y;
        p.h[2] = (_Float16)a.z;  p.h[3] = (_Float16)a.w;
        p.h[4] = (_Float16)b.x;  p.h[5] = (_Float16)b.y;
        p.h[6] = (_Float16)b.z;  p.h[7] = (_Float16)b.w;
        *(int4*)&ldsB[buf][pofs] = p.u;                   // ds_store_b128
    };

    union Frag { int4 i4[2]; v16h h; };

    v8f acc0 = {}, acc1 = {}, acc2 = {}, acc3 = {};

    produce(0, 0);
    __syncthreads();
    for (int c = 0; c < NCHUNK; ++c) {
        const int cur = c & 1;
        if (c + 1 < NCHUNK) {
            __builtin_prefetch(wrow + (c + 2) * 32, 0, 1);  // global_prefetch_b8
            produce(cur ^ 1, c + 1);
        }
        // A fragment (global, row-major f16)
        Frag ua;
        ua.i4[0] = *(const int4*)(arow + c * 32 + kA0);     // global b128
        ua.i4[1] = *(const int4*)(arow + c * 32 + kA0 + 16);

        // All four B fragments loaded up-front into distinct live ranges so
        // the scheduler can clause 8x ds_load_b128 and use partial dscnt waits.
        const int4* bp = (const int4*)&ldsB[cur][lane * 32];
        Frag ub0, ub1, ub2, ub3;
        ub0.i4[0] = bp[0];    ub0.i4[1] = bp[1];            // ds_load_b128 x8
        ub1.i4[0] = bp[64];   ub1.i4[1] = bp[65];
        ub2.i4[0] = bp[128];  ub2.i4[1] = bp[129];
        ub3.i4[0] = bp[192];  ub3.i4[1] = bp[193];

        acc0 = __builtin_amdgcn_wmma_f32_16x16x32_f16(
                   false, ua.h, false, ub0.h, (short)0, acc0, false, false);
        acc1 = __builtin_amdgcn_wmma_f32_16x16x32_f16(
                   false, ua.h, false, ub1.h, (short)0, acc1, false, false);
        acc2 = __builtin_amdgcn_wmma_f32_16x16x32_f16(
                   false, ua.h, false, ub2.h, (short)0, acc2, false, false);
        acc3 = __builtin_amdgcn_wmma_f32_16x16x32_f16(
                   false, ua.h, false, ub3.h, (short)0, acc3, false, false);

        __syncthreads();   // one barrier per chunk (double-buffered)
    }

    // ---- split-K combine: atomic f32 accumulate into tbuf[kk][m][n]
    float* tb = tbuf + (size_t)kk * BATCH * ND;
    const int mofs = Mbase + ((lane & 16) ? 8 : 0);
    const int nb   = n0 + (lane & 15);
    #pragma unroll
    for (int r = 0; r < 8; ++r) {
        float* rowp = tb + (size_t)(mofs + r) * ND + nb;
        atomicAdd(rowp +  0, acc0[r]);
        atomicAdd(rowp + 16, acc1[r]);
        atomicAdd(rowp + 32, acc2[r]);
        atomicAdd(rowp + 48, acc3[r]);
    }
}

// -------- kernel 5: out = bc + flag * sum_k selu(t[k])*w2[n][k] ------------
__global__ void tn_epilogue(const float* __restrict__ tbuf,
                            const float* __restrict__ w2,
                            const float* __restrict__ bc,
                            const float* __restrict__ flag,
                            float*       __restrict__ out) {
    const float la = 1.0507009873554805f;   // SELU lambda
    const float al = 1.6732632423543772f;   // SELU alpha
    int idx = blockIdx.x * blockDim.x + threadIdx.x;   // b*4096 + n
    int n = idx & (ND - 1);
    float s = 0.f;
    #pragma unroll
    for (int kk = 0; kk < 3; ++kk) {
        float v = tbuf[(size_t)kk * BATCH * ND + idx];
        v = (v > 0.f) ? la * v : la * al * (__expf(v) - 1.f);
        s += v * w2[n * 3 + kk];
    }
    out[idx] = bc[n] + s * flag[n];
}

// ---------------------------------------------------------------------------
extern "C" void kernel_launch(void* const* d_in, const int* in_sizes, int n_in,
                              void* d_out, int out_size, void* d_ws, size_t ws_size,
                              hipStream_t stream) {
    const float* x     = (const float*)d_in[0];   // [128][4096]
    const float* bc    = (const float*)d_in[1];   // [4096]
    const float* flag  = (const float*)d_in[2];   // [4096]
    const int*   Brows = (const int*)  d_in[3];   // [NNZ]
    const int*   Bcols = (const int*)  d_in[4];   // [NNZ]
    const float* Bvals = (const float*)d_in[5];   // [NNZ]
    const float* w1    = (const float*)d_in[6];   // [3][4096][4096]
    const float* w2    = (const float*)d_in[7];   // [4096][3]
    float* out = (float*)d_out;

    const int total = BATCH * ND;                 // 524288
    // workspace layout (contiguous f32 unless noted):
    float*    xT   = (float*)d_ws;                // [4096][128]      2 MB
    float*    yT   = xT + total;                  // [4096][128]      2 MB
    float*    tbuf = yT + total;                  // [3][128][4096]   6 MB
    _Float16* xh   = (_Float16*)(tbuf + 3 * total); // [128][4096]    1 MB

    // zero yT + tbuf in one contiguous sweep (4 * total floats)
    tn_zero<<<total / 256, 256, 0, stream>>>((float4*)yT, total);
    tn_transpose<<<total / 256, 256, 0, stream>>>(x, xT);
    tn_spmv<<<(NNZ * 32) / 256, 256, 0, stream>>>(Brows, Bcols, Bvals, xT, yT);
    tn_prep<<<total / 256, 256, 0, stream>>>(yT, bc, flag, xh);
    tn_gemm_splitk<<<(ND / NTILE) * 3 * KSPLIT, 256, 0, stream>>>(xh, w1, tbuf);
    tn_epilogue<<<total / 256, 256, 0, stream>>>(tbuf, w2, bc, flag, out);
}
